// WannAttention_62740882260038
// MI455X (gfx1250) — matrix operand order repaired and checked by reading, not compiled, not run on hardware
//
#include <hip/hip_runtime.h>
#include <hip/hip_bf16.h>

// ---------------------------------------------------------------------------
// Problem constants (reference: B=2, S=2048, D=2048, H=16, HD=128)
// ---------------------------------------------------------------------------
constexpr int Bc  = 2;
constexpr int Sc  = 2048;
constexpr int Dc  = 2048;
constexpr int Hc  = 16;
constexpr int HDc = 128;
constexpr int Mc  = Bc * Sc;        // 4096 projection rows

// ---------------------------------------------------------------------------
// WMMA types (gfx1250, wave32): v_wmma_f32_16x16x32_bf16
// ---------------------------------------------------------------------------
typedef __attribute__((ext_vector_type(16))) __bf16        v16bf;
typedef __attribute__((ext_vector_type(8)))  float         v8f;
typedef __attribute__((ext_vector_type(4)))  unsigned int  v4u;   // 16B chunk

union AF {                 // A operand: 16 bf16, two contiguous 16B halves
    v16bf v;
    v4u   u[2];
};

__device__ __forceinline__ v8f wmma_bf16(const v16bf& a, const v16bf& b, v8f c) {
    return __builtin_amdgcn_wmma_f32_16x16x32_bf16(
        false, a, false, b, (short)0, c, false, false);
}

// fp32 -> bf16 (round to nearest even)
__device__ __forceinline__ unsigned short f2bf(float f) {
    unsigned int u = __float_as_uint(f);
    unsigned int r = u + 0x7FFFu + ((u >> 16) & 1u);
    return (unsigned short)(r >> 16);
}
__device__ __forceinline__ unsigned int pack_bf2(float a, float b) {
    return (unsigned int)f2bf(a) | ((unsigned int)f2bf(b) << 16);
}

// ---------------------------------------------------------------------------
// CDNA5 async global->LDS copies (ASYNCcnt) with synchronous fallback.
// Builtin signature (probed): (AS1 int4*, AS3 int4*, imm offset, imm cpol).
// ---------------------------------------------------------------------------
#if defined(__has_builtin)
#if __has_builtin(__builtin_amdgcn_global_load_async_to_lds_b128) && \
    __has_builtin(__builtin_amdgcn_s_wait_asynccnt)
#define HAS_ASYNC_CP 1
#endif
#endif
#ifndef HAS_ASYNC_CP
#define HAS_ASYNC_CP 0
#endif

typedef int gv4i __attribute__((vector_size(16)));
typedef __attribute__((address_space(1))) gv4i* as1_v4i_ptr;
typedef __attribute__((address_space(3))) gv4i* as3_v4i_ptr;

__device__ __forceinline__ void async_cp16(const unsigned short* g, unsigned short* l) {
#if HAS_ASYNC_CP
    __builtin_amdgcn_global_load_async_to_lds_b128(
        (as1_v4i_ptr)(g), (as3_v4i_ptr)(l), 0, 0);
#else
    *(v4u*)l = *(const v4u*)g;
#endif
}

#if HAS_ASYNC_CP
#define ASYNC_WAIT(n) __builtin_amdgcn_s_wait_asynccnt(n)
#else
#define ASYNC_WAIT(n)
#endif

// ---------------------------------------------------------------------------
// Kernel 0: fp32 -> bf16 conversion (optionally scaled by shared_weight).
// ---------------------------------------------------------------------------
__global__ __launch_bounds__(256) void convert_bf16_kernel(
    const float* __restrict__ src, unsigned short* __restrict__ dst,
    const float* __restrict__ sw_p, int apply_sw)
{
    float s = apply_sw ? sw_p[0] : 1.0f;
    size_t i = ((size_t)blockIdx.x * 256 + threadIdx.x) * 8;
    float4 a = *(const float4*)(src + i);
    float4 b = *(const float4*)(src + i + 4);
    v4u o;
    o.x = pack_bf2(a.x * s, a.y * s);
    o.y = pack_bf2(a.z * s, a.w * s);
    o.z = pack_bf2(b.x * s, b.y * s);
    o.w = pack_bf2(b.z * s, b.w * s);
    *(v4u*)(dst + i) = o;
}

// ---------------------------------------------------------------------------
// Shared GEMM core: C[128x256] += A[128xK] * B[256xK]^T, A/B bf16 row-major
// (K contiguous).  256 threads = 8 waves in 2x4; wave tile 64x64 = 4x4 WMMA.
// Double-buffered LDS tiles filled by async global->LDS copies.
// ---------------------------------------------------------------------------
__device__ __forceinline__ void stage_gemm(
    const unsigned short* __restrict__ A, const unsigned short* __restrict__ B,
    unsigned short* la, unsigned short* lb, int m0, int n0, int k0, int tid)
{
    #pragma unroll
    for (int i = 0; i < 2; i++) {                       // A tile: 128x32 = 512x16B
        int idx = tid + i * 256, row = idx >> 2, c = idx & 3;
        async_cp16(A + (size_t)(m0 + row) * Dc + k0 + c * 8, la + row * 32 + c * 8);
    }
    #pragma unroll
    for (int i = 0; i < 4; i++) {                       // B tile: 256x32 = 1024x16B
        int idx = tid + i * 256, row = idx >> 2, c = idx & 3;
        async_cp16(B + (size_t)(n0 + row) * Dc + k0 + c * 8, lb + row * 32 + c * 8);
    }
}

__device__ __forceinline__ void gemm_core_128x256(
    const unsigned short* __restrict__ A, const unsigned short* __restrict__ B,
    int m0, int n0, unsigned short* lA, unsigned short* lB, v8f acc[4][4])
{
    const int tid  = threadIdx.x;
    const int wave = tid >> 5, lane = tid & 31;
    const int half = lane >> 4, lrow = lane & 15;
    const int wm   = wave >> 2, wn = wave & 3;

    stage_gemm(A, B, lA, lB, m0, n0, 0, tid);           // prologue: buffer 0

    for (int k0 = 0, it = 0; k0 < Dc; k0 += 32, ++it) {
        const int cur = it & 1;
        const bool has_next = (k0 + 32) < Dc;
        if (has_next)
            stage_gemm(A, B, lA + (cur ^ 1) * (128 * 32), lB + (cur ^ 1) * (256 * 32),
                       m0, n0, k0 + 32, tid);
        if (has_next) { ASYNC_WAIT(6); } else { ASYNC_WAIT(0); }
        __syncthreads();                                 // current tiles visible

        const unsigned short* sA = lA + cur * (128 * 32);
        const unsigned short* sB = lB + cur * (256 * 32);

        // B fragments: 32 contiguous bytes per lane -> direct v16bf loads
        v16bf b[4];
        #pragma unroll
        for (int nt = 0; nt < 4; nt++) {
            int c = wn * 64 + nt * 16 + lrow;
            b[nt] = *(const v16bf*)(sB + c * 32 + half * 16);
        }
        // mt-outer: one live A fragment at a time (smaller live set)
        #pragma unroll
        for (int mt = 0; mt < 4; mt++) {
            AF a;
            int r = wm * 64 + mt * 16 + lrow;
            a.u[0] = *(const v4u*)(sA + r * 32 + half * 16);
            a.u[1] = *(const v4u*)(sA + r * 32 + half * 16 + 8);
            #pragma unroll
            for (int nt = 0; nt < 4; nt++)
                acc[mt][nt] = wmma_bf16(a.v, b[nt], acc[mt][nt]);
        }
        __syncthreads();                                 // before next-next staging
    }
}

// ---------------------------------------------------------------------------
// Kernel 1: fused Q/K/V projection from bf16 x and bf16 (mask*sw) weights.
// blockIdx.z = 0/1/2 -> q/k/v.  Q,K stored [B,H,S,HD] bf16; V stored
// transposed [B,H,HD,S] bf16.  head_mask applied at the epilogue.
// ---------------------------------------------------------------------------
__global__ __launch_bounds__(256) void proj_qkv_kernel(
    const unsigned short* __restrict__ xb,
    const unsigned short* __restrict__ wq, const unsigned short* __restrict__ wk,
    const unsigned short* __restrict__ wv,
    const float* __restrict__ head_mask,
    unsigned short* __restrict__ qb, unsigned short* __restrict__ kb,
    unsigned short* __restrict__ vtb)
{
    __shared__ __align__(16) unsigned short lA[2 * 128 * 32];
    __shared__ __align__(16) unsigned short lB[2 * 256 * 32];

    const unsigned short* W = (blockIdx.z == 0) ? wq : (blockIdx.z == 1) ? wk : wv;
    const int m0 = blockIdx.y * 128;
    const int n0 = blockIdx.x * 256;
    const int tid  = threadIdx.x;
    const int wave = tid >> 5, lane = tid & 31;
    const int half = lane >> 4, lrow = lane & 15;
    const int wm   = wave >> 2, wn = wave & 3;

    v8f acc[4][4] = {};
    gemm_core_128x256(xb, W, m0, n0, lA, lB, acc);

    #pragma unroll
    for (int mt = 0; mt < 4; mt++) {
        #pragma unroll
        for (int nt = 0; nt < 4; nt++) {
            int Ng = n0 + wn * 64 + nt * 16 + lrow;
            int h  = Ng >> 7, hd = Ng & (HDc - 1);
            float scale = head_mask[h];
            #pragma unroll
            for (int r = 0; r < 8; r++) {
                int mg = m0 + wm * 64 + mt * 16 + r + 8 * half;
                int bI = mg >> 11, sI = mg & (Sc - 1);
                unsigned short val = f2bf(acc[mt][nt][r] * scale);
                if (blockIdx.z == 2) {
                    vtb[((size_t)((bI * Hc + h) * HDc + hd)) * Sc + sI] = val;
                } else {
                    unsigned short* dst = (blockIdx.z == 0) ? qb : kb;
                    dst[((size_t)((bI * Hc + h) * Sc + sI)) * HDc + hd] = val;
                }
            }
        }
    }
}

// ---------------------------------------------------------------------------
// Kernel 2: flash attention, online softmax, causal + additive mask.
// 128 threads = 4 waves; block = 64 query rows of one (b,h); wave = 16 rows.
// K [32][128] and V^T [128][32] tiles double-buffered in LDS via async copies.
// ---------------------------------------------------------------------------
__device__ __forceinline__ void stage_kv(
    const unsigned short* __restrict__ kbase, const unsigned short* __restrict__ vbase,
    unsigned short* lk, unsigned short* lv, int kk0, int tid)
{
    #pragma unroll
    for (int i = 0; i < 4; i++) {                        // K tile: 512 x 16B
        int idx = tid + i * 128, row = idx >> 4, c4 = idx & 15;
        async_cp16(kbase + (size_t)(kk0 + row) * HDc + c4 * 8, lk + row * 128 + c4 * 8);
    }
    #pragma unroll
    for (int i = 0; i < 4; i++) {                        // V^T tile: 512 x 16B
        int idx = tid + i * 128, row = idx >> 2, c4 = idx & 3;
        async_cp16(vbase + (size_t)row * Sc + kk0 + c4 * 8, lv + row * 32 + c4 * 8);
    }
}

__global__ __launch_bounds__(128) void flash_attn_kernel(
    const unsigned short* __restrict__ qb, const unsigned short* __restrict__ kb,
    const unsigned short* __restrict__ vtb, const float* __restrict__ amask,
    unsigned short* __restrict__ aout)
{
    __shared__ __align__(16) unsigned short lK[2][32 * 128];   // [key][hd]
    __shared__ __align__(16) unsigned short lV[2][128 * 32];   // [hd][key]
    __shared__ __align__(16) unsigned short lP[4][16 * 32];    // per-wave P staging

    const int bh   = blockIdx.y;
    const int bI   = bh >> 4;                 // H = 16
    const int h    = bh & 15;
    const int q0   = blockIdx.x * 64;
    const int tid  = threadIdx.x;
    const int wave = tid >> 5, lane = tid & 31;
    const int half = lane >> 4, lrow = lane & 15;
    const int qr0  = q0 + wave * 16;
    const float scale = 0.08838834764831845f; // HD^-0.5

    const unsigned short* kbase = kb  + (size_t)bh * Sc * HDc;
    const unsigned short* vbase = vtb + (size_t)bh * HDc * Sc;

    // Q fragments (A layout) for this wave's 16 rows, kept in registers
    AF qf[4];
    {
        const v4u* qp = (const v4u*)(qb + ((size_t)bh * Sc + qr0 + lrow) * HDc);
        #pragma unroll
        for (int kc = 0; kc < 4; kc++) {
            qf[kc].u[0] = qp[kc * 4 + 2 * half];
            qf[kc].u[1] = qp[kc * 4 + 2 * half + 1];
        }
    }

    float mi[8], li[8];
    #pragma unroll
    for (int r = 0; r < 8; r++) { mi[r] = -3.0e38f; li[r] = 0.0f; }
    v8f o[8] = {};

    const int njb = (q0 + 64) >> 5;           // causal: key blocks of 32
    stage_kv(kbase, vbase, lK[0], lV[0], 0, tid);

    for (int jb = 0; jb < njb; jb++) {
        const int cur = jb & 1;
        const int kk0 = jb * 32;
        const bool has_next = (jb + 1) < njb;
        if (has_next) stage_kv(kbase, vbase, lK[cur ^ 1], lV[cur ^ 1], kk0 + 32, tid);
        if (has_next) { ASYNC_WAIT(8); } else { ASYNC_WAIT(0); }
        __syncthreads();

        // S = Q K^T  (two 16-key sub-tiles, K-dim 128 in 4 steps)
        v8f st[2] = {};
        #pragma unroll
        for (int kc = 0; kc < 4; kc++) {
            #pragma unroll
            for (int nt = 0; nt < 2; nt++) {
                int kr = nt * 16 + lrow;
                v16bf bk = *(const v16bf*)(lK[cur] + kr * 128 + kc * 32 + half * 16);
                st[nt] = wmma_bf16(qf[kc].v, bk, st[nt]);
            }
        }

        // scale + causal + additive attention mask
        float am0 = amask[bI * Sc + kk0 + lrow];
        float am1 = amask[bI * Sc + kk0 + 16 + lrow];
        float sc[2][8];
        #pragma unroll
        for (int nt = 0; nt < 2; nt++) {
            int key = kk0 + nt * 16 + lrow;
            float add = (1.0f - (nt ? am1 : am0)) * -10000.0f;
            #pragma unroll
            for (int r = 0; r < 8; r++) {
                int qrow = qr0 + r + 8 * half;
                float v = st[nt][r] * scale + add;
                sc[nt][r] = (key > qrow) ? -3.0e38f : v;
            }
        }

        // online softmax: each row lives across the 16 lanes of a half
        float pr[2][8];
        #pragma unroll
        for (int r = 0; r < 8; r++) {
            float mx = fmaxf(sc[0][r], sc[1][r]);
            #pragma unroll
            for (int d = 1; d <= 8; d <<= 1) mx = fmaxf(mx, __shfl_xor(mx, d, 32));
            float nm   = fmaxf(mi[r], mx);
            float corr = __expf(mi[r] - nm);
            float rs = 0.0f;
            #pragma unroll
            for (int nt = 0; nt < 2; nt++) {
                float p = __expf(sc[nt][r] - nm);
                pr[nt][r] = p; rs += p;
            }
            #pragma unroll
            for (int d = 1; d <= 8; d <<= 1) rs += __shfl_xor(rs, d, 32);
            li[r] = li[r] * corr + rs;
            mi[r] = nm;
            #pragma unroll
            for (int t = 0; t < 8; t++) o[t][r] *= corr;
        }

        // P: C layout -> LDS -> A layout fragment
        unsigned short* pw = lP[wave];
        #pragma unroll
        for (int nt = 0; nt < 2; nt++)
            #pragma unroll
            for (int r = 0; r < 8; r++)
                pw[(r + 8 * half) * 32 + nt * 16 + lrow] = f2bf(pr[nt][r]);
        __syncthreads();

        AF pf;
        pf.u[0] = *(const v4u*)(lP[wave] + lrow * 32 + half * 16);
        pf.u[1] = *(const v4u*)(lP[wave] + lrow * 32 + half * 16 + 8);

        // O += P @ V  (8 hd sub-tiles)
        #pragma unroll
        for (int nt = 0; nt < 8; nt++) {
            int vr = nt * 16 + lrow;
            v16bf bv = *(const v16bf*)(lV[cur] + vr * 32 + half * 16);
            o[nt] = wmma_bf16(pf.v, bv, o[nt]);
        }
        __syncthreads();                       // all reads of buf `cur` done
    }

    // finalize: divide by row sums, store bf16 [B,S,D] for the out-projection
    #pragma unroll
    for (int r = 0; r < 8; r++) {
        float inv = 1.0f / li[r];
        #pragma unroll
        for (int nt = 0; nt < 8; nt++) o[nt][r] *= inv;
    }
    #pragma unroll
    for (int nt = 0; nt < 8; nt++) {
        int d = h * HDc + nt * 16 + lrow;
        #pragma unroll
        for (int r = 0; r < 8; r++) {
            int srow = qr0 + r + 8 * half;
            aout[((size_t)(bI * Sc + srow)) * Dc + d] = f2bf(o[nt][r]);
        }
    }
}

// ---------------------------------------------------------------------------
// Kernel 3: output projection, bf16 attention output x bf16 (out_mask*sw),
// fp32 result straight to d_out.
// ---------------------------------------------------------------------------
__global__ __launch_bounds__(256) void proj_out_kernel(
    const unsigned short* __restrict__ attn, const unsigned short* __restrict__ wo,
    float* __restrict__ out)
{
    __shared__ __align__(16) unsigned short lA[2 * 128 * 32];
    __shared__ __align__(16) unsigned short lB[2 * 256 * 32];

    const int m0 = blockIdx.y * 128;
    const int n0 = blockIdx.x * 256;
    const int tid  = threadIdx.x;
    const int wave = tid >> 5, lane = tid & 31;
    const int half = lane >> 4, lrow = lane & 15;
    const int wm   = wave >> 2, wn = wave & 3;

    v8f acc[4][4] = {};
    gemm_core_128x256(attn, wo, m0, n0, lA, lB, acc);

    #pragma unroll
    for (int mt = 0; mt < 4; mt++) {
        #pragma unroll
        for (int nt = 0; nt < 4; nt++) {
            int Ng = n0 + wn * 64 + nt * 16 + lrow;
            #pragma unroll
            for (int r = 0; r < 8; r++) {
                int mg = m0 + wm * 64 + mt * 16 + r + 8 * half;
                out[(size_t)mg * Dc + Ng] = acc[mt][nt][r];
            }
        }
    }
}

// ---------------------------------------------------------------------------
// Launch: convert -> proj_qkv -> flash_attn -> proj_out (stream-ordered).
// Workspace (bf16): x | wq | wk | wv | wo | q | k | v^T | attn_out  (~118 MB)
// ---------------------------------------------------------------------------
extern "C" void kernel_launch(void* const* d_in, const int* in_sizes, int n_in,
                              void* d_out, int out_size, void* d_ws, size_t ws_size,
                              hipStream_t stream) {
    (void)in_sizes; (void)n_in; (void)out_size; (void)ws_size;
    const float* x  = (const float*)d_in[0];
    const float* qm = (const float*)d_in[1];
    const float* km = (const float*)d_in[2];
    const float* vm = (const float*)d_in[3];
    const float* om = (const float*)d_in[4];
    const float* hm = (const float*)d_in[5];
    const float* am = (const float*)d_in[6];
    const float* sw = (const float*)d_in[7];
    float* out = (float*)d_out;

    const size_t nx = (size_t)Mc * Dc;        // 8.4M  (x / per-tensor q,k,v,attn)
    const size_t nw = (size_t)Dc * Dc;        // 4.2M  (each weight)
    unsigned short* xb  = (unsigned short*)d_ws;
    unsigned short* wq  = xb  + nx;
    unsigned short* wk  = wq  + nw;
    unsigned short* wv  = wk  + nw;
    unsigned short* wo  = wv  + nw;
    unsigned short* qb  = wo  + nw;
    unsigned short* kb  = qb  + nx;
    unsigned short* vtb = kb  + nx;
    unsigned short* ab  = vtb + nx;

    convert_bf16_kernel<<<dim3(nx / 2048), 256, 0, stream>>>(x,  xb, sw, 0);
    convert_bf16_kernel<<<dim3(nw / 2048), 256, 0, stream>>>(qm, wq, sw, 1);
    convert_bf16_kernel<<<dim3(nw / 2048), 256, 0, stream>>>(km, wk, sw, 1);
    convert_bf16_kernel<<<dim3(nw / 2048), 256, 0, stream>>>(vm, wv, sw, 1);
    convert_bf16_kernel<<<dim3(nw / 2048), 256, 0, stream>>>(om, wo, sw, 1);

    dim3 g1(Dc / 256, Mc / 128, 3);
    proj_qkv_kernel<<<g1, 256, 0, stream>>>(xb, wq, wk, wv, hm, qb, kb, vtb);

    dim3 g2(Sc / 64, Bc * Hc);
    flash_attn_kernel<<<g2, 128, 0, stream>>>(qb, kb, vtb, am, ab);

    dim3 g3(Dc / 256, Mc / 128);
    proj_out_kernel<<<g3, 256, 0, stream>>>(ab, wo, out);
}
